// text2embedding_model_New_66984309948613
// MI455X (gfx1250) — matrix operand order repaired and checked by reading, not compiled, not run on hardware
//
#include <hip/hip_runtime.h>

typedef __attribute__((ext_vector_type(16))) __bf16 v16bf;
typedef __attribute__((ext_vector_type(8)))  __bf16 v8bf;
typedef __attribute__((ext_vector_type(8)))  float  v8f;
typedef __attribute__((ext_vector_type(4)))  int    v4i;

#define HID   1024
#define N3    3072
#define BATCH 128
#define TIN   64
#define TOUT  90
#define EMB   300
#define EMBP  320   /* EMB padded to multiple of 32 */
#define POSE  514
#define POSEP 528   /* POSE padded to multiple of 16 */
#define KC_ENC (EMBP + HID)   /* 1344 */
#define KC_DEC (HID + HID)    /* 2048 */

static __device__ __forceinline__ __bf16 f2bf(float f) {
  unsigned u = __builtin_bit_cast(unsigned, f);
  unsigned r = (u + 0x7FFFu + ((u >> 16) & 1u)) >> 16;
  return __builtin_bit_cast(__bf16, (unsigned short)r);
}

// ---- gfx1250 async global->LDS copy (ASYNCcnt path), with fallback --------
static __device__ __forceinline__ void async_copy_b128(const char* gsrc, char* ldst) {
#if __has_builtin(__builtin_amdgcn_global_load_async_to_lds_b128)
  __builtin_amdgcn_global_load_async_to_lds_b128(
      (__attribute__((address_space(1))) v4i*)gsrc,
      (__attribute__((address_space(3))) v4i*)ldst, 0, 0);
#else
  *(v8bf*)ldst = *(const v8bf*)gsrc;
#endif
}

static __device__ __forceinline__ void wait_async_copies() {
#if __has_builtin(__builtin_amdgcn_global_load_async_to_lds_b128)
#if __has_builtin(__builtin_amdgcn_s_wait_asynccnt)
  __builtin_amdgcn_s_wait_asynccnt(0);
#else
  asm volatile("s_wait_asynccnt 0x0" ::: "memory");
#endif
#endif
}

// ---- WMMA tile loaders (contiguous-K order in A and B, wave32) -----------------------
// A (16x32 MxK bf16): lane l -> row m0+(l&15); halves 0..15 = K khi*16..+15
// where khi = l>>4.
static __device__ __forceinline__ v16bf load_a_tile(const __bf16* __restrict__ base,
                                                    int ld, int m0, int k0, int lane) {
  int m   = m0 + (lane & 15);
  int khi = lane >> 4;
  const __bf16* p = base + (size_t)m * ld + k0 + khi * 16;
  v8bf lo = *(const v8bf*)(p);
  v8bf hi = *(const v8bf*)(p + 8);
  v16bf a;
#pragma unroll
  for (int i = 0; i < 8; ++i) { a[i] = lo[i]; a[8 + i] = hi[i]; }
  return a;
}

// B (32x16 KxN bf16): lane l -> col (l&15) of a 16-row strip [16 x ld],
// 16 contiguous K halves starting at k0+(l>>4)*16. `strip` may be LDS.
static __device__ __forceinline__ v16bf load_b_strip(const __bf16* strip,
                                                     int ld, int k0, int lane) {
  int n = lane & 15;
  int koff = k0 + (lane >> 4) * 16;
  return *(const v16bf*)(strip + (size_t)n * ld + koff);
}

static __device__ __forceinline__ v8f wmma_bf16(v16bf a, v16bf b, v8f c) {
  return __builtin_amdgcn_wmma_f32_16x16x32_bf16(false, a, false, b, (short)0, c,
                                                 false, false);
}

// ---- setup kernels --------------------------------------------------------
__global__ void pack_w_kernel(const float* __restrict__ src, __bf16* __restrict__ dst,
                              int N, int K, int Kseg, int dstLD, int colOff) {
  long i = (long)blockIdx.x * 256 + threadIdx.x;
  long total = (long)N * Kseg;
  if (i >= total) return;
  int n = (int)(i / Kseg), k = (int)(i % Kseg);
  dst[(size_t)n * dstLD + colOff + k] = (k < K) ? f2bf(src[(size_t)n * K + k]) : f2bf(0.f);
}

__global__ void pack_fc_kernel(const float* __restrict__ src, __bf16* __restrict__ dst) {
  int i = blockIdx.x * 256 + threadIdx.x;
  if (i >= POSEP * HID) return;
  int n = i / HID, k = i % HID;
  dst[i] = (n < POSE) ? f2bf(src[(size_t)n * HID + k]) : f2bf(0.f);
}

// x_emb[t][b][k] = enc_emb[in_text[b][t]][k]  (time-major, K padded to 320)
__global__ void gather_enc_kernel(const float* __restrict__ emb,
                                  const int* __restrict__ text,
                                  __bf16* __restrict__ dst) {
  long i = (long)blockIdx.x * 256 + threadIdx.x;
  long total = (long)TIN * BATCH * EMBP;
  if (i >= total) return;
  int k = (int)(i % EMBP);
  int row = (int)(i / EMBP);
  int b = row % BATCH, t = row / BATCH;
  int tok = text[b * TIN + t];
  dst[i] = (k < EMB) ? f2bf(emb[(size_t)tok * EMB + k]) : f2bf(0.f);
}

// dec_x[t][b][k] = dec_emb[poses[b][max(t-1,0)]][k]
__global__ void gather_dec_kernel(const float* __restrict__ emb,
                                  const int* __restrict__ poses,
                                  __bf16* __restrict__ dst) {
  long i = (long)blockIdx.x * 256 + threadIdx.x;
  long total = (long)TOUT * BATCH * HID;
  if (i >= total) return;
  int k = (int)(i % HID);
  int row = (int)(i / HID);
  int b = row % BATCH, t = row / BATCH;
  int ts = (t == 0) ? 0 : (t - 1);
  int tok = poses[b * TOUT + ts];
  dst[i] = f2bf(emb[(size_t)tok * HID + k]);
}

__global__ void init_h_kernel(float* f1, float* f2, __bf16* b1, __bf16* b2) {
  int i = blockIdx.x * 256 + threadIdx.x;
  if (i >= BATCH * HID) return;
  f1[i] = 0.f; f2[i] = 0.f; b1[i] = f2bf(0.f); b2[i] = f2bf(0.f);
}

__global__ void add_h0_kernel(const float* __restrict__ a, const float* __restrict__ b,
                              float* __restrict__ hf, __bf16* __restrict__ hb) {
  int i = blockIdx.x * 256 + threadIdx.x;
  if (i >= BATCH * HID) return;
  float v = a[i] + b[i];
  hf[i] = v; hb[i] = f2bf(v);
}

// ---- fused GRU step -------------------------------------------------------
// One "job" = one GRU direction's step. gridDim.y selects job (encoder fuses
// both directions into a single launch). grid.x = 64 j-tiles of 16 hidden
// cols; block = 128 (4 waves x 32 rows, M register-blocked x2).
// The WG's 3 gate weight strips (rows j0.., HID+j0.., 2HID+j0.., each
// 16 x kc, contiguous in the packed W) are staged into LDS once via async
// global->LDS copies; the K loop then reads B from LDS with no barriers,
// so weights are read exactly once per step from L2.
struct GruJob {
  const __bf16* X;
  const __bf16* Hbf;
  const float*  Hf;
  const __bf16* W;
  const float*  bih;
  const float*  bhh;
  float*        Hf_out;
  __bf16*       Hbf_out;
  __bf16*       HS_out;   // may be null
};

__global__ __launch_bounds__(128)
void gru_step_kernel(GruJob jobA, GruJob jobB, int xld, int kc) {
  extern __shared__ char smem_raw[];
  __bf16* smem = (__bf16*)smem_raw;
  const GruJob j = (blockIdx.y == 0) ? jobA : jobB;

  const int tid  = threadIdx.x;
  const int lane = tid & 31;
  const int wave = tid >> 5;
  const int m0 = wave * 32;           // two 16-row tiles: m0, m0+16
  const int j0 = blockIdx.x * 16;
  const int nkx = xld / 32;           // K-tiles in x region
  const int nkh = HID / 32;           // K-tiles in h region
  const size_t stripElems = (size_t)16 * kc;       // bf16 elems per gate strip
  const size_t stripBytes = stripElems * 2;        // multiple of 2048

  // ---- stage 3 gate weight strips into LDS (async, ASYNCcnt) ----
#pragma unroll
  for (int g = 0; g < 3; ++g) {
    const char* src = (const char*)(j.W + (size_t)(g * HID + j0) * kc);
    char* dst = smem_raw + (size_t)g * stripBytes;
    for (size_t ofs = (size_t)tid * 16; ofs < stripBytes; ofs += 128 * 16)
      async_copy_b128(src + ofs, dst + ofs);
  }
  wait_async_copies();
  __syncthreads();

  const __bf16* sR = smem;
  const __bf16* sZ = smem + stripElems;
  const __bf16* sN = smem + 2 * stripElems;

  v8f accR0 = {}, accR1 = {}, accZ0 = {}, accZ1 = {};
  v8f accN0 = {}, accN1 = {};         // i_n during x-phase
  v8f accH0 = {}, accH1 = {};         // h_n during h-phase

  // ---- x phase: K = [0, xld) ----
  for (int kt = 0; kt < nkx; ++kt) {
    const int k0 = kt * 32;
    v16bf a0 = load_a_tile(j.X, xld, m0,      k0, lane);
    v16bf a1 = load_a_tile(j.X, xld, m0 + 16, k0, lane);
    v16bf bR = load_b_strip(sR, kc, k0, lane);
    v16bf bZ = load_b_strip(sZ, kc, k0, lane);
    v16bf bN = load_b_strip(sN, kc, k0, lane);
    accR0 = wmma_bf16(a0, bR, accR0);
    accR1 = wmma_bf16(a1, bR, accR1);
    accZ0 = wmma_bf16(a0, bZ, accZ0);
    accZ1 = wmma_bf16(a1, bZ, accZ1);
    accN0 = wmma_bf16(a0, bN, accN0);
    accN1 = wmma_bf16(a1, bN, accN1);
  }
  // ---- h phase: K = [xld, kc) ----
  for (int kt = 0; kt < nkh; ++kt) {
    const int k0h = kt * 32;          // offset within h state
    const int k0w = xld + k0h;        // offset within combined strip
    v16bf a0 = load_a_tile(j.Hbf, HID, m0,      k0h, lane);
    v16bf a1 = load_a_tile(j.Hbf, HID, m0 + 16, k0h, lane);
    v16bf bR = load_b_strip(sR, kc, k0w, lane);
    v16bf bZ = load_b_strip(sZ, kc, k0w, lane);
    v16bf bN = load_b_strip(sN, kc, k0w, lane);
    accR0 = wmma_bf16(a0, bR, accR0);
    accR1 = wmma_bf16(a1, bR, accR1);
    accZ0 = wmma_bf16(a0, bZ, accZ0);
    accZ1 = wmma_bf16(a1, bZ, accZ1);
    accH0 = wmma_bf16(a0, bN, accH0);
    accH1 = wmma_bf16(a1, bN, accH1);
  }

  const int jl  = j0 + (lane & 15);
  const int khi = lane >> 4;
  const float br  = j.bih[jl] + j.bhh[jl];
  const float bz  = j.bih[HID + jl] + j.bhh[HID + jl];
  const float bin = j.bih[2 * HID + jl];
  const float bhn = j.bhh[2 * HID + jl];
#pragma unroll
  for (int p = 0; p < 2; ++p) {
    const v8f aR = p ? accR1 : accR0;
    const v8f aZ = p ? accZ1 : accZ0;
    const v8f aN = p ? accN1 : accN0;
    const v8f aH = p ? accH1 : accH0;
    const int mb = m0 + p * 16;
#pragma unroll
    for (int r = 0; r < 8; ++r) {
      const int m = mb + r + 8 * khi;
      float rg = 1.0f / (1.0f + __expf(-(aR[r] + br)));
      float zg = 1.0f / (1.0f + __expf(-(aZ[r] + bz)));
      float ng = tanhf(aN[r] + bin + rg * (aH[r] + bhn));
      float hp = j.Hf[(size_t)m * HID + jl];
      float hn = (1.0f - zg) * ng + zg * hp;
      j.Hf_out[(size_t)m * HID + jl] = hn;
      __bf16 hb = f2bf(hn);
      j.Hbf_out[(size_t)m * HID + jl] = hb;
      if (j.HS_out) j.HS_out[(size_t)m * HID + jl] = hb;
    }
  }
}

// ---- output projection: out[t,b,:] = hs @ fc_W^T + fc_b --------------------
// grid = (90 m-blocks of 128 rows, 33 n-tiles of 16 cols), block = 128
// (4 waves x 32 rows, M register-blocked x2).
__global__ __launch_bounds__(128)
void proj_kernel(const __bf16* __restrict__ HS,
                 const __bf16* __restrict__ Wfc,
                 const float* __restrict__ fcb,
                 float* __restrict__ out) {
  const int lane = threadIdx.x & 31;
  const int wave = threadIdx.x >> 5;
  const int m0 = blockIdx.x * 128 + wave * 32;
  const int n0 = blockIdx.y * 16;
  const __bf16* strip = Wfc + (size_t)n0 * HID;
  v8f acc0 = {}, acc1 = {};
  for (int kt = 0; kt < HID / 32; ++kt) {
    const int k0 = kt * 32;
    v16bf a0 = load_a_tile(HS, HID, m0,      k0, lane);
    v16bf a1 = load_a_tile(HS, HID, m0 + 16, k0, lane);
    v16bf b  = load_b_strip(strip, HID, k0, lane);
    acc0 = wmma_bf16(a0, b, acc0);
    acc1 = wmma_bf16(a1, b, acc1);
  }
  const int o = n0 + (lane & 15);
  const int khi = lane >> 4;
  if (o < POSE) {
    const float bb = fcb[o];
#pragma unroll
    for (int p = 0; p < 2; ++p) {
      const v8f acc = p ? acc1 : acc0;
      const int mb = m0 + p * 16;
#pragma unroll
      for (int r = 0; r < 8; ++r) {
        const int m = mb + r + 8 * khi;
        out[(size_t)m * POSE + o] = acc[r] + bb;
      }
    }
  }
}

// ---------------------------------------------------------------------------
extern "C" void kernel_launch(void* const* d_in, const int* in_sizes, int n_in,
                              void* d_out, int out_size, void* d_ws, size_t ws_size,
                              hipStream_t stream) {
  (void)in_sizes; (void)n_in; (void)out_size; (void)ws_size;
  const float* enc_emb   = (const float*)d_in[0];
  const float* enc_Wih_f = (const float*)d_in[1];
  const float* enc_Whh_f = (const float*)d_in[2];
  const float* enc_bih_f = (const float*)d_in[3];
  const float* enc_bhh_f = (const float*)d_in[4];
  const float* enc_Wih_b = (const float*)d_in[5];
  const float* enc_Whh_b = (const float*)d_in[6];
  const float* enc_bih_b = (const float*)d_in[7];
  const float* enc_bhh_b = (const float*)d_in[8];
  const float* dec_emb   = (const float*)d_in[9];
  const float* dec_Wih   = (const float*)d_in[10];
  const float* dec_Whh   = (const float*)d_in[11];
  const float* dec_bih   = (const float*)d_in[12];
  const float* dec_bhh   = (const float*)d_in[13];
  const float* fc_W      = (const float*)d_in[14];
  const float* fc_b      = (const float*)d_in[15];
  const int*   in_text   = (const int*)d_in[16];
  const int*   poses     = (const int*)d_in[18];
  float* out = (float*)d_out;

  char* ws = (char*)d_ws;
  size_t off = 0;
  auto alloc = [&](size_t bytes) -> char* {
    char* p = ws + off;
    off += (bytes + 255) & ~(size_t)255;
    return p;
  };

  __bf16* WcEncF = (__bf16*)alloc((size_t)N3 * KC_ENC * 2);
  __bf16* WcEncB = (__bf16*)alloc((size_t)N3 * KC_ENC * 2);
  __bf16* WcDec  = (__bf16*)alloc((size_t)N3 * KC_DEC * 2);
  __bf16* Wfc    = (__bf16*)alloc((size_t)POSEP * HID * 2);
  __bf16* Xenc   = (__bf16*)alloc((size_t)TIN * BATCH * EMBP * 2);
  __bf16* Xdec   = (__bf16*)alloc((size_t)TOUT * BATCH * HID * 2);
  __bf16* HS     = (__bf16*)alloc((size_t)TOUT * BATCH * HID * 2);
  float*  encF_f32[2]; __bf16* encF_bf[2];
  float*  encB_f32[2]; __bf16* encB_bf[2];
  float*  dec_f32[2];  __bf16* dec_bf[2];
  for (int p = 0; p < 2; ++p) {
    encF_f32[p] = (float*)alloc((size_t)BATCH * HID * 4);
    encF_bf[p]  = (__bf16*)alloc((size_t)BATCH * HID * 2);
    encB_f32[p] = (float*)alloc((size_t)BATCH * HID * 4);
    encB_bf[p]  = (__bf16*)alloc((size_t)BATCH * HID * 2);
    dec_f32[p]  = (float*)alloc((size_t)BATCH * HID * 4);
    dec_bf[p]   = (__bf16*)alloc((size_t)BATCH * HID * 2);
  }

  auto g1 = [](long n) { return (unsigned)((n + 255) / 256); };

  // weight pack (fp32 -> bf16, K-concatenated [Wih | Whh])
  pack_w_kernel<<<g1((long)N3 * EMBP), 256, 0, stream>>>(enc_Wih_f, WcEncF, N3, EMB, EMBP, KC_ENC, 0);
  pack_w_kernel<<<g1((long)N3 * HID),  256, 0, stream>>>(enc_Whh_f, WcEncF, N3, HID, HID, KC_ENC, EMBP);
  pack_w_kernel<<<g1((long)N3 * EMBP), 256, 0, stream>>>(enc_Wih_b, WcEncB, N3, EMB, EMBP, KC_ENC, 0);
  pack_w_kernel<<<g1((long)N3 * HID),  256, 0, stream>>>(enc_Whh_b, WcEncB, N3, HID, HID, KC_ENC, EMBP);
  pack_w_kernel<<<g1((long)N3 * HID),  256, 0, stream>>>(dec_Wih, WcDec, N3, HID, HID, KC_DEC, 0);
  pack_w_kernel<<<g1((long)N3 * HID),  256, 0, stream>>>(dec_Whh, WcDec, N3, HID, HID, KC_DEC, HID);
  pack_fc_kernel<<<g1((long)POSEP * HID), 256, 0, stream>>>(fc_W, Wfc);

  // embedding gathers (time-major, bf16)
  gather_enc_kernel<<<g1((long)TIN * BATCH * EMBP), 256, 0, stream>>>(enc_emb, in_text, Xenc);
  gather_dec_kernel<<<g1((long)TOUT * BATCH * HID), 256, 0, stream>>>(dec_emb, poses, Xdec);

  // h0 = 0 for both encoder directions (parity-0 buffers)
  init_h_kernel<<<g1((long)BATCH * HID), 256, 0, stream>>>(encF_f32[0], encB_f32[0], encF_bf[0], encB_bf[0]);

  const size_t smem_enc = (size_t)3 * 16 * KC_ENC * 2;  // 129 KB
  const size_t smem_dec = (size_t)3 * 16 * KC_DEC * 2;  // 192 KB

  // encoder recurrence: both directions fused in one launch (gridDim.y = 2)
  for (int t = 0; t < TIN; ++t) {
    int pi = t & 1, po = (t + 1) & 1;
    GruJob jf = { Xenc + (size_t)t * BATCH * EMBP, encF_bf[pi], encF_f32[pi],
                  WcEncF, enc_bih_f, enc_bhh_f,
                  encF_f32[po], encF_bf[po], nullptr };
    GruJob jb = { Xenc + (size_t)t * BATCH * EMBP, encB_bf[pi], encB_f32[pi],
                  WcEncB, enc_bih_b, enc_bhh_b,
                  encB_f32[po], encB_bf[po], nullptr };
    gru_step_kernel<<<dim3(HID / 16, 2), 128, smem_enc, stream>>>(jf, jb, EMBP, KC_ENC);
  }

  // dec_h0 = hf + hb   (TIN even -> final enc state is in parity-0 buffers)
  add_h0_kernel<<<g1((long)BATCH * HID), 256, 0, stream>>>(encF_f32[0], encB_f32[0], dec_f32[0], dec_bf[0]);

  // decoder recurrence, streaming hidden states into HS
  for (int t = 0; t < TOUT; ++t) {
    int pi = t & 1, po = (t + 1) & 1;
    GruJob jd = { Xdec + (size_t)t * BATCH * HID, dec_bf[pi], dec_f32[pi],
                  WcDec, dec_bih, dec_bhh,
                  dec_f32[po], dec_bf[po], HS + (size_t)t * BATCH * HID };
    gru_step_kernel<<<dim3(HID / 16, 1), 128, smem_dec, stream>>>(jd, jd, HID, KC_DEC);
  }

  // out[t,b,o] = hs[t,b,:] . fc_W[o,:] + fc_b[o]
  proj_kernel<<<dim3(TOUT, POSEP / 16), 128, 0, stream>>>(HS, Wfc, fc_b, out);
}
